// MultiHeadAttention_44976897524606
// MI455X (gfx1250) — matrix operand order repaired and checked by reading, not compiled, not run on hardware
//
#include <hip/hip_runtime.h>
#include <hip/hip_bf16.h>

#define B_  16
#define S_  512
#define D_  768
#define H_  12
#define DKV 64
#define NEGV (-1e9f)

typedef __attribute__((ext_vector_type(16))) _Float16 v16h;
typedef __attribute__((ext_vector_type(8)))  float    v8f;
typedef __attribute__((ext_vector_type(4)))  float    v4f;
typedef __attribute__((ext_vector_type(4)))  _Float16 h4;

// ---------------------------------------------------------------------------
// WMMA wrapper: D = A(16x32 f16) * B(32x16 f16) + C(16x16 f32)
// ---------------------------------------------------------------------------
__device__ __forceinline__ v8f wmma32(v16h a, v16h b, v8f c) {
    return __builtin_amdgcn_wmma_f32_16x16x32_f16(
        /*neg_a=*/false, a, /*neg_b=*/false, b,
        /*c_mod=*/(short)0, c, /*reuse_a=*/false, /*reuse_b=*/false);
}

// ---------------------------------------------------------------------------
// Fragment loaders (this code's contiguous k order, same for A and B).
// A 16x32 f16: lane L -> row M=L%16; half j -> K = (L/16)*16 + j
// B 32x16 f16: lane L -> col N=L%16; half j -> K = (L/16)*16 + j
// ---------------------------------------------------------------------------
__device__ __forceinline__ v16h frag_a_f16(const _Float16* __restrict__ X, int ld,
                                           int m0, int kb) {
    const int lane = threadIdx.x & 31;
    const _Float16* row = X + (size_t)(m0 + (lane & 15)) * ld + kb;
    const int grp = (lane >> 4) << 4;
    v16h a;
#pragma unroll
    for (int blk = 0; blk < 4; ++blk) {
        const int k = grp + (blk << 2);
        h4 x = *(const h4*)(row + k);
#pragma unroll
        for (int t = 0; t < 4; ++t) a[blk * 4 + t] = x[t];
    }
    return a;
}

// B = X^T where X is row-major [N,K] (f16): weights W[n,k], k[S,64]
__device__ __forceinline__ v16h frag_bT_f16(const _Float16* __restrict__ X, int ld,
                                            int n0, int kb) {
    const int lane = threadIdx.x & 31;
    const _Float16* p = X + (size_t)(n0 + (lane & 15)) * ld + kb + ((lane >> 4) << 4);
    v16h b;
#pragma unroll
    for (int blk = 0; blk < 4; ++blk) {
        h4 x = *(const h4*)(p + blk * 4);
#pragma unroll
        for (int t = 0; t < 4; ++t) b[blk * 4 + t] = x[t];
    }
    return b;
}

// B = tile^T from an LDS-resident [64][32] f16 tile (ds_load path)
__device__ __forceinline__ v16h frag_b_lds(const _Float16* tile, int n0) {
    const int lane = threadIdx.x & 31;
    const _Float16* p = tile + (n0 + (lane & 15)) * 32 + ((lane >> 4) << 4);
    v16h b;
#pragma unroll
    for (int blk = 0; blk < 4; ++blk) {
        h4 x = *(const h4*)(p + blk * 4);
#pragma unroll
        for (int t = 0; t < 4; ++t) b[blk * 4 + t] = x[t];
    }
    return b;
}

// ===========================================================================
// Kernel 0: bulk f32 -> f16 conversion (vectorized x4)
// ===========================================================================
__global__ __launch_bounds__(256) void k_cvt(const float* __restrict__ src,
                                             _Float16* __restrict__ dst, int n4) {
    const int i = blockIdx.x * 256 + threadIdx.x;
    if (i < n4) {
        v4f x = ((const v4f*)src)[i];
        h4 y;
        y[0] = (_Float16)x.x; y[1] = (_Float16)x.y;
        y[2] = (_Float16)x.z; y[3] = (_Float16)x.w;
        ((h4*)dst)[i] = y;
    }
}

// ===========================================================================
// Kernel 1: QKV projection. Each wave: 16x64 strip (one head's columns).
// C[8192,768] = Xh[8192,768] * Wh^T + bias ; q,k -> [B,H,S,64], v -> [B,H,64,S]
// ===========================================================================
__global__ __launch_bounds__(128) void k_qkv(
    const _Float16* __restrict__ Qh, const _Float16* __restrict__ Kh,
    const _Float16* __restrict__ Vh,
    const _Float16* __restrict__ Wqh, const float* __restrict__ bq,
    const _Float16* __restrict__ Wkh, const float* __restrict__ bk,
    const _Float16* __restrict__ Wvh, const float* __restrict__ bv,
    _Float16* __restrict__ qb, _Float16* __restrict__ kb,
    _Float16* __restrict__ vbT) {
    const int mt = blockIdx.x * 4 + threadIdx.y;   // 512 row tiles (B*S/16)
    const int h  = blockIdx.y;                     // 12 heads (64-col strip)
    const int which = blockIdx.z;                  // 0=q 1=k 2=v
    const _Float16* X = (which == 0) ? Qh : (which == 1) ? Kh : Vh;
    const _Float16* W = (which == 0) ? Wqh : (which == 1) ? Wkh : Wvh;
    const float* bias = (which == 0) ? bq : (which == 1) ? bk : bv;

    v8f acc0 = {}, acc1 = {}, acc2 = {}, acc3 = {};
    for (int kk = 0; kk < D_; kk += 32) {
        const v16h a = frag_a_f16(X, D_, mt * 16, kk);
        acc0 = wmma32(a, frag_bT_f16(W, D_, h * 64 +  0, kk), acc0);
        acc1 = wmma32(a, frag_bT_f16(W, D_, h * 64 + 16, kk), acc1);
        acc2 = wmma32(a, frag_bT_f16(W, D_, h * 64 + 32, kk), acc2);
        acc3 = wmma32(a, frag_bT_f16(W, D_, h * 64 + 48, kk), acc3);
    }
    v8f accs[4] = {acc0, acc1, acc2, acc3};

    const int lane = threadIdx.x & 31;
#pragma unroll
    for (int t = 0; t < 4; ++t) {
        const int dk = t * 16 + (lane & 15);
        const float bvv = bias[h * 64 + dk];
#pragma unroll
        for (int r = 0; r < 8; ++r) {
            const int m = mt * 16 + r + ((lane >> 4) << 3);
            const int bidx = m / S_, s = m - bidx * S_;
            const _Float16 val = (_Float16)(accs[t][r] + bvv);
            if (which == 2)        // v transposed: [B,H,64,S]
                vbT[(((size_t)bidx * H_ + h) * DKV + dk) * S_ + s] = val;
            else if (which == 1)   // k: [B,H,S,64]
                kb[((((size_t)bidx * H_ + h) * S_) + s) * DKV + dk] = val;
            else                   // q: [B,H,S,64]
                qb[((((size_t)bidx * H_ + h) * S_) + s) * DKV + dk] = val;
        }
    }
}

// ===========================================================================
// Kernel 2: scores (q.k^T/8) + mask + help blend + softmax.
// One wave per (b,h,16-row q tile); 16x512 f32 staged in LDS for softmax.
// attn -> f32 d_out and f16 workspace.
// ===========================================================================
__global__ __launch_bounds__(32) void k_scores(
    const _Float16* __restrict__ qb, const _Float16* __restrict__ kb,
    const unsigned char* __restrict__ amask, const float* __restrict__ help,
    float* __restrict__ attn_out, _Float16* __restrict__ attn_h) {
    __shared__ float sc[16][S_ + 4];
    const int qt = blockIdx.x;   // 32
    const int h  = blockIdx.y;   // 12
    const int b  = blockIdx.z;   // 16
    const int lane = threadIdx.x & 31;

    const _Float16* qrow = qb + (((size_t)b * H_ + h) * S_) * DKV;
    const _Float16* krow = kb + (((size_t)b * H_ + h) * S_) * DKV;
    const size_t hb = ((((size_t)b * H_ + h) * S_) + qt * 16) * S_;  // [B,H,S,S]
    const size_t ab = (((size_t)b * S_) + qt * 16) * S_;             // [B,S,S]

    __builtin_prefetch(help + hb + lane * 64, 0, 0);   // global_prefetch_b8

    const v16h a0 = frag_a_f16(qrow, DKV, qt * 16, 0);
    const v16h a1 = frag_a_f16(qrow, DKV, qt * 16, 32);

    for (int ntile = 0; ntile < 32; ++ntile) {
        v8f acc = {};
        acc = wmma32(a0, frag_bT_f16(krow, DKV, ntile * 16, 0), acc);
        acc = wmma32(a1, frag_bT_f16(krow, DKV, ntile * 16, 32), acc);
        const int n = ntile * 16 + (lane & 15);
#pragma unroll
        for (int r = 0; r < 8; ++r) {
            const int m = r + ((lane >> 4) << 3);
            float s = acc[r] * 0.125f;                       // 1/sqrt(64)
            if (amask[ab + (size_t)m * S_ + n]) s = NEGV;
            const float hs = help[hb + (size_t)m * S_ + n];
            if (hs > -1e8f) s = 0.5f * hs + 0.5f * s;
            sc[m][n] = s;
        }
    }
    __syncthreads();

    for (int r = 0; r < 16; ++r) {
        float vals[16];
        float mx = -3.4e38f;
#pragma unroll
        for (int c = 0; c < 16; ++c) {
            const float x = sc[r][lane + 32 * c];
            vals[c] = x;
            mx = fmaxf(mx, x);
        }
        for (int off = 16; off; off >>= 1) mx = fmaxf(mx, __shfl_xor(mx, off, 32));
        float sum = 0.f;
#pragma unroll
        for (int c = 0; c < 16; ++c) {
            const float e = __expf(vals[c] - mx);
            vals[c] = e;
            sum += e;
        }
        for (int off = 16; off; off >>= 1) sum += __shfl_xor(sum, off, 32);
        const float inv = 1.0f / sum;
#pragma unroll
        for (int c = 0; c < 16; ++c) {
            const float p = vals[c] * inv;
            const size_t o = hb + (size_t)r * S_ + lane + 32 * c;
            attn_out[o] = p;
            attn_h[o] = (_Float16)p;
        }
    }
}

// ===========================================================================
// Kernel 3: context = attn[S,S] x v[S,64] per (b,h). Wave: 16x64 strip.
// The B-matrix strip (Vt 64x32 tile, 4KB) is shared by all 4 waves of the
// block -> stage it in LDS once via double-buffered async global->LDS copies
// (global_load_async_to_lds_b128 + s_wait_asynccnt pipelining), then all
// waves read B fragments with ds loads.  -> f16 ctx [B,S,768]
// ===========================================================================
__global__ __launch_bounds__(128) void k_ctx(
    const _Float16* __restrict__ attn_h, const _Float16* __restrict__ vbT,
    _Float16* __restrict__ ctx) {
    __shared__ _Float16 vt_tile[2][DKV][32];      // 2 x 4KB double buffer
    const int mt = blockIdx.x * 4 + threadIdx.y;  // 32 (S/16)
    const int bh = blockIdx.y;                    // 192
    const int b = bh / H_, h = bh - b * H_;
    const _Float16* A  = attn_h + ((size_t)bh * S_) * S_;
    const _Float16* Vt = vbT + ((size_t)bh * DKV) * S_;   // [64, S]
    const int tid = threadIdx.y * 32 + threadIdx.x;       // 0..127

    // stage one 64x32 tile (256 x 16B segments, 2 per thread) into LDS[buf]
    auto stage = [&](int kk, int buf) {
#pragma unroll
        for (int i = 0; i < 2; ++i) {
            const int seg = tid + i * 128;
            const int row = seg >> 2;             // 0..63 (dv)
            const int c   = (seg & 3) << 3;       // 0,8,16,24 (halves)
            const unsigned lds_off =
                (unsigned)(uintptr_t)&vt_tile[buf][row][c];
            const unsigned long long gaddr =
                (unsigned long long)(uintptr_t)(Vt + (size_t)row * S_ + kk + c);
            asm volatile("global_load_async_to_lds_b128 %0, %1, off"
                         :: "v"(lds_off), "v"(gaddr) : "memory");
        }
    };

    v8f acc0 = {}, acc1 = {}, acc2 = {}, acc3 = {};
    stage(0, 0);
    for (int step = 0; step < 16; ++step) {
        const int cur = step & 1;
        if (step < 15) {
            stage((step + 1) * 32, cur ^ 1);                  // prefetch next
            asm volatile("s_wait_asynccnt 0x2" ::: "memory"); // cur done, next in flight
        } else {
            asm volatile("s_wait_asynccnt 0x0" ::: "memory");
        }
        __syncthreads();
        const v16h a = frag_a_f16(A, S_, mt * 16, step * 32);
        const _Float16* tile = &vt_tile[cur][0][0];
        acc0 = wmma32(a, frag_b_lds(tile,  0), acc0);
        acc1 = wmma32(a, frag_b_lds(tile, 16), acc1);
        acc2 = wmma32(a, frag_b_lds(tile, 32), acc2);
        acc3 = wmma32(a, frag_b_lds(tile, 48), acc3);
        __syncthreads();   // tile consumed before it is overwritten
    }
    v8f accs[4] = {acc0, acc1, acc2, acc3};

    const int lane = threadIdx.x & 31;
#pragma unroll
    for (int t = 0; t < 4; ++t) {
        const int n = t * 16 + (lane & 15);
#pragma unroll
        for (int r = 0; r < 8; ++r) {
            const int s = mt * 16 + r + ((lane >> 4) << 3);
            ctx[(((size_t)b * S_) + s) * D_ + h * DKV + n] = (_Float16)accs[t][r];
        }
    }
}

// ===========================================================================
// Kernel 4: out_pre = ctx[8192,768] x Wo^T + bo + residual(Q). Wave: 16x64.
// ===========================================================================
__global__ __launch_bounds__(128) void k_oproj(
    const _Float16* __restrict__ ctx, const _Float16* __restrict__ Woh,
    const float* __restrict__ bo, const float* __restrict__ Qin,
    float* __restrict__ outpre) {
    const int mt = blockIdx.x * 4 + threadIdx.y;  // 512
    const int nt = blockIdx.y;                    // 12 (64-col strips)
    v8f acc0 = {}, acc1 = {}, acc2 = {}, acc3 = {};
    for (int kk = 0; kk < D_; kk += 32) {
        const v16h a = frag_a_f16(ctx, D_, mt * 16, kk);
        acc0 = wmma32(a, frag_bT_f16(Woh, D_, nt * 64 +  0, kk), acc0);
        acc1 = wmma32(a, frag_bT_f16(Woh, D_, nt * 64 + 16, kk), acc1);
        acc2 = wmma32(a, frag_bT_f16(Woh, D_, nt * 64 + 32, kk), acc2);
        acc3 = wmma32(a, frag_bT_f16(Woh, D_, nt * 64 + 48, kk), acc3);
    }
    v8f accs[4] = {acc0, acc1, acc2, acc3};

    const int lane = threadIdx.x & 31;
#pragma unroll
    for (int t = 0; t < 4; ++t) {
        const int n = nt * 64 + t * 16 + (lane & 15);
        const float bvv = bo[n];
#pragma unroll
        for (int r = 0; r < 8; ++r) {
            const size_t m = (size_t)mt * 16 + r + ((lane >> 4) << 3);
            outpre[m * D_ + n] = accs[t][r] + bvv + Qin[m * D_ + n];
        }
    }
}

// ===========================================================================
// Kernel 5: row LayerNorm (768 elems/row), 256 threads -> 3 elems/thread
// ===========================================================================
__global__ __launch_bounds__(256) void k_ln(
    const float* __restrict__ xin, const float* __restrict__ gamma,
    const float* __restrict__ beta, float* __restrict__ out) {
    const int row = blockIdx.x;
    const float* x = xin + (size_t)row * D_;
    const int t = threadIdx.x, wid = t >> 5, lane = t & 31;
    __shared__ float redS[8], redQ[8];

    const float v0 = x[t], v1 = x[t + 256], v2 = x[t + 512];
    float s = v0 + v1 + v2;
    float q = v0 * v0 + v1 * v1 + v2 * v2;
    for (int off = 16; off; off >>= 1) {
        s += __shfl_xor(s, off, 32);
        q += __shfl_xor(q, off, 32);
    }
    if (lane == 0) { redS[wid] = s; redQ[wid] = q; }
    __syncthreads();
    float ts = 0.f, tq = 0.f;
#pragma unroll
    for (int i = 0; i < 8; ++i) { ts += redS[i]; tq += redQ[i]; }
    const float mu = ts * (1.0f / D_);
    const float var = tq * (1.0f / D_) - mu * mu;
    const float rstd = rsqrtf(var + 1e-5f);

    float* o = out + (size_t)row * D_;
#pragma unroll
    for (int i = 0; i < 3; ++i) {
        const int c = t + i * 256;
        o[c] = (x[c] - mu) * rstd * gamma[c] + beta[c];
    }
}

// ===========================================================================
extern "C" void kernel_launch(void* const* d_in, const int* in_sizes, int n_in,
                              void* d_out, int out_size, void* d_ws, size_t ws_size,
                              hipStream_t stream) {
    const float* Q    = (const float*)d_in[0];
    const float* K    = (const float*)d_in[1];
    const float* V    = (const float*)d_in[2];
    const unsigned char* amask = (const unsigned char*)d_in[3];  // bool8
    const float* help = (const float*)d_in[4];
    const float* Wq   = (const float*)d_in[5];
    const float* bq   = (const float*)d_in[6];
    const float* Wk   = (const float*)d_in[7];
    const float* bk   = (const float*)d_in[8];
    const float* Wv   = (const float*)d_in[9];
    const float* bv   = (const float*)d_in[10];
    const float* Wo   = (const float*)d_in[11];
    const float* bo   = (const float*)d_in[12];
    const float* gamma= (const float*)d_in[13];
    const float* beta = (const float*)d_in[14];

    float* out_main = (float*)d_out;                       // [B,S,D]
    float* attn_out = out_main + (size_t)B_ * S_ * D_;     // [B,H,S,S]

    char* w = (char*)d_ws;
    size_t off = 0;
    auto carve = [&](size_t bytes) -> void* {
        void* p = w + off;
        off = (off + bytes + 255) & ~(size_t)255;
        return p;
    };
    const size_t nAct = (size_t)B_ * S_ * D_;        // 6.29M
    const size_t nW   = (size_t)D_ * D_;             // 0.59M
    const size_t nQKV = (size_t)B_ * H_ * S_ * DKV;  // 6.29M
    _Float16* Qh  = (_Float16*)carve(nAct * 2);
    _Float16* Kh  = (_Float16*)carve(nAct * 2);
    _Float16* Vh  = (_Float16*)carve(nAct * 2);
    _Float16* Wqh = (_Float16*)carve(nW * 2);
    _Float16* Wkh = (_Float16*)carve(nW * 2);
    _Float16* Wvh = (_Float16*)carve(nW * 2);
    _Float16* Woh = (_Float16*)carve(nW * 2);
    _Float16* qb  = (_Float16*)carve(nQKV * 2);
    _Float16* kb  = (_Float16*)carve(nQKV * 2);
    _Float16* vbT = (_Float16*)carve(nQKV * 2);
    _Float16* attn_h = (_Float16*)carve((size_t)B_ * H_ * S_ * S_ * 2);
    _Float16* ctx    = (_Float16*)carve(nAct * 2);
    float*    outpre = (float*)carve(nAct * 4);
    (void)ws_size; (void)n_in; (void)in_sizes; (void)out_size;

    // ---- stage 0: convert activations + weights to f16 (one pass) ----
    const int a4 = (int)(nAct / 4), w4 = (int)(nW / 4);
    const int gA = (a4 + 255) / 256, gW = (w4 + 255) / 256;
    k_cvt<<<gA, 256, 0, stream>>>(Q,  Qh,  a4);
    k_cvt<<<gA, 256, 0, stream>>>(K,  Kh,  a4);
    k_cvt<<<gA, 256, 0, stream>>>(V,  Vh,  a4);
    k_cvt<<<gW, 256, 0, stream>>>(Wq, Wqh, w4);
    k_cvt<<<gW, 256, 0, stream>>>(Wk, Wkh, w4);
    k_cvt<<<gW, 256, 0, stream>>>(Wv, Wvh, w4);
    k_cvt<<<gW, 256, 0, stream>>>(Wo, Woh, w4);

    // ---- stage 1: QKV projections (WMMA) ----
    dim3 g1(B_ * S_ / 16 / 4, H_, 3), blk4(32, 4);
    k_qkv<<<g1, blk4, 0, stream>>>(Qh, Kh, Vh, Wqh, bq, Wkh, bk, Wvh, bv,
                                   qb, kb, vbT);

    // ---- stage 2: scores + mask/help blend + softmax (WMMA + LDS) ----
    dim3 g2(S_ / 16, H_, B_);
    k_scores<<<g2, 32, 0, stream>>>(qb, kb, amask, help, attn_out, attn_h);

    // ---- stage 3: context GEMM (WMMA + async global->LDS B staging) ----
    dim3 g3(S_ / 16 / 4, B_ * H_);
    k_ctx<<<g3, blk4, 0, stream>>>(attn_h, vbT, ctx);

    // ---- stage 4: output projection + residual (WMMA) ----
    dim3 g4(B_ * S_ / 16 / 4, D_ / DKV);
    k_oproj<<<g4, blk4, 0, stream>>>(ctx, Woh, bo, Q, outpre);

    // ---- stage 5: LayerNorm ----
    k_ln<<<B_ * S_, 256, 0, stream>>>(outpre, gamma, beta, out_main);
}
